// Attention_82257213653665
// MI455X (gfx1250) — compile-verified, matrix operand order checked by reading, not run on hardware
//
#include <hip/hip_runtime.h>
#include <hip/hip_bf16.h>

typedef __bf16 bf16_t;
typedef __attribute__((ext_vector_type(16))) __bf16 v16bf;
typedef __attribute__((ext_vector_type(8)))  __bf16 v8bf;
typedef __attribute__((ext_vector_type(8)))  float  v8f;
typedef __attribute__((ext_vector_type(4)))  float  v4f;

#define NHEADS  16
#define DH      64
#define BQ      64      // q rows per block (16 per wave x 4 waves)
#define BK      64      // k rows per iteration
#define SQ      2048
#define SK      2048
#define CD      1024    // channels = NHEADS*DH
#define NW      4       // waves per block
#define NT      (SK / BK)
#define SCALE   0.125f  // 1/sqrt(64)
#define NEGB    999999.0f

#define LDT     80      // padded LDS row stride for async tiles (bf16; 160B, 16B-aligned)
#define LDP     72      // padded LDS row stride for P staging / f32-path tiles
#define LDK     72
#define LDV     72

#if __has_builtin(__builtin_amdgcn_global_load_async_to_lds_b128)
#define HAVE_ASYNC 1
#else
#define HAVE_ASYNC 0
#endif

// Builtin's parameter types (from the round-2 diagnostic): generic int4* src,
// LDS-qualified int4* dst.
typedef int gv4i __attribute__((__vector_size__(4 * sizeof(int))));
typedef __attribute__((address_space(3))) gv4i lds_gv4i;

__device__ __forceinline__ gv4i* glb_v4(const void* p) {
    return (gv4i*)(unsigned long long)p;  // integer round-trip strips const
}
__device__ __forceinline__ lds_gv4i* lds_v4(void* p) {
    // generic LDS address low 32 bits == LDS byte offset (ISA aperture rule)
    return (lds_gv4i*)(unsigned int)(unsigned long long)p;
}

union V16u { v16bf v; v8bf h[2]; };

// Assemble a v16bf WMMA operand (A-row or B-column) from a bf16 LDS row.
// CDNA5 16-bit operand layout: lanes<16 hold K=[0..7,16..23] of the 32-chunk,
// lanes>=16 hold K=[8..15,24..31]  ->  reads at cbase+8*hi and cbase+16+8*hi.
__device__ __forceinline__ v16bf lds_chunk(const bf16_t* row, int cbase, int hi) {
    V16u u;
    u.h[0] = *reinterpret_cast<const v8bf*>(row + cbase + 8 * hi);
    u.h[1] = *reinterpret_cast<const v8bf*>(row + cbase + 16 + 8 * hi);
    return u.v;
}

// Same layout, sourced from global f32 (used for the persistent Q operand).
__device__ __forceinline__ v16bf gmem_f32_chunk(const float* row, int cbase, int hi) {
    const float* p0 = row + cbase + 8 * hi;
    const float* p1 = row + cbase + 16 + 8 * hi;
    v4f x0 = *reinterpret_cast<const v4f*>(p0);
    v4f x1 = *reinterpret_cast<const v4f*>(p0 + 4);
    v4f x2 = *reinterpret_cast<const v4f*>(p1);
    v4f x3 = *reinterpret_cast<const v4f*>(p1 + 4);
    V16u u;
#pragma unroll
    for (int i = 0; i < 4; ++i) {
        u.h[0][i]     = (bf16_t)x0[i];
        u.h[0][i + 4] = (bf16_t)x1[i];
        u.h[1][i]     = (bf16_t)x2[i];
        u.h[1][i + 4] = (bf16_t)x3[i];
    }
    return u.v;
}

// ---------------------------------------------------------------------------
// Pre-pass 1: elementwise f32 -> bf16 for K (one-time; main loop re-reads K
// 32x per (b,h) from L2, so converting once halves L2 bytes and removes VALU).
// ---------------------------------------------------------------------------
__global__ __launch_bounds__(256)
void convert_k_bf16(const float* __restrict__ K, bf16_t* __restrict__ Kb) {
    union { bf16_t h[8]; uint4 u; } pk;
    long i = ((long)blockIdx.x * blockDim.x + threadIdx.x) * 8;
    v4f a = *reinterpret_cast<const v4f*>(K + i);
    v4f b = *reinterpret_cast<const v4f*>(K + i + 4);
#pragma unroll
    for (int t = 0; t < 4; ++t) { pk.h[t] = (bf16_t)a[t]; pk.h[t + 4] = (bf16_t)b[t]; }
    *reinterpret_cast<uint4*>(Kb + i) = pk.u;
}

// ---------------------------------------------------------------------------
// Pre-pass 2: V -> per-head transposed bf16  Vtw[b][h][d][Sk]
// (makes the PV B-operand row-contiguous; kills per-tile LDS transpose).
// ---------------------------------------------------------------------------
__global__ __launch_bounds__(128)
void transpose_v_bf16(const float* __restrict__ V, bf16_t* __restrict__ Vtw) {
    __shared__ __align__(16) bf16_t T[DH * LDP];  // [d][s] staging
    const int kt = blockIdx.x, h = blockIdx.y, b = blockIdx.z, tid = threadIdx.x;
    const float* Vg = V + ((long)b * SK + kt * BK) * CD + h * DH;
#pragma unroll
    for (int i = 0; i < 8; ++i) {
        int chunk = i * 128 + tid;
        int r  = chunk >> 4;          // s row 0..63
        int c4 = (chunk & 15) * 4;    // d col base
        v4f vd = *reinterpret_cast<const v4f*>(Vg + (long)r * CD + c4);
#pragma unroll
        for (int t = 0; t < 4; ++t) T[(c4 + t) * LDP + r] = (bf16_t)vd[t];
    }
    __syncthreads();
    bf16_t* dst = Vtw + ((long)(b * NHEADS + h)) * DH * SK + kt * BK;  // row stride SK
#pragma unroll
    for (int i = 0; i < 4; ++i) {
        int chunk = i * 128 + tid;
        int r = chunk >> 3;           // d row 0..63
        int c = (chunk & 7) * 8;      // s col base
        *reinterpret_cast<uint4*>(dst + (long)r * SK + c) =
            *reinterpret_cast<const uint4*>(&T[r * LDP + c]);
    }
}

// ---------------------------------------------------------------------------
// Main flash-attention kernel, bf16 tiles, async-to-LDS double buffering.
// ---------------------------------------------------------------------------
__global__ __launch_bounds__(128)
void fattn_wmma_bf16(const float* __restrict__ Qf, const bf16_t* __restrict__ Kb,
                     const bf16_t* __restrict__ Vtw,
                     const unsigned char* __restrict__ qmask,
                     const unsigned char* __restrict__ vmask,
                     float* __restrict__ Out) {
    __shared__ __align__(16) bf16_t KtS[2][BK * LDT];   // K tile, row = key idx
    __shared__ __align__(16) bf16_t VtS[2][DH * LDT];   // V^T tile, row = d idx
    __shared__ __align__(16) bf16_t Pb[NW * 16 * LDP];  // per-wave P staging

    const int qt   = blockIdx.x;
    const int h    = blockIdx.y;
    const int b    = blockIdx.z;
    const int tid  = threadIdx.x;
    const int wave = tid >> 5;
    const int lane = tid & 31;
    const int ln   = lane & 15;
    const int hi   = lane >> 4;

    const int q0 = qt * BQ + wave * 16;

    const bf16_t* Ksrc = Kb + (long)b * SK * CD + h * DH;            // row stride CD
    const bf16_t* Vsrc = Vtw + ((long)(b * NHEADS + h)) * DH * SK;   // row stride SK

    // Issue one K tile + one V^T tile into LDS buffer `buf`.
    auto issue_tile = [&](int kt, int buf) {
#pragma unroll
        for (int i = 0; i < 4; ++i) {
            int chunk = i * 128 + tid;        // 512 x 16B chunks per tile
            int r = chunk >> 3;               // row 0..63
            int c = (chunk & 7) * 8;          // col base (bf16 elems)
#if HAVE_ASYNC
            __builtin_amdgcn_global_load_async_to_lds_b128(
                glb_v4(Ksrc + (long)(kt * BK + r) * CD + c),
                lds_v4(&KtS[buf][r * LDT + c]), 0, 0);
            __builtin_amdgcn_global_load_async_to_lds_b128(
                glb_v4(Vsrc + (long)r * SK + kt * BK + c),
                lds_v4(&VtS[buf][r * LDT + c]), 0, 0);
#else
            uint4 kd = *reinterpret_cast<const uint4*>(Ksrc + (long)(kt * BK + r) * CD + c);
            uint4 vd = *reinterpret_cast<const uint4*>(Vsrc + (long)r * SK + kt * BK + c);
            *reinterpret_cast<uint4*>(&KtS[buf][r * LDT + c]) = kd;
            *reinterpret_cast<uint4*>(&VtS[buf][r * LDT + c]) = vd;
#endif
        }
    };

    issue_tile(0, 0);

    // Persistent Q operand (f32 -> bf16 once; overlaps with the first async tile).
    const long qoff = ((long)b * SQ + (q0 + ln)) * CD + h * DH;
    v16bf aq[2];
    aq[0] = gmem_f32_chunk(Qf + qoff, 0, hi);
    aq[1] = gmem_f32_chunk(Qf + qoff, 32, hi);

    v8f o[4];
    float m[8], l[8];
#pragma unroll
    for (int dk = 0; dk < 4; ++dk)
#pragma unroll
        for (int r = 0; r < 8; ++r) o[dk][r] = 0.0f;
#pragma unroll
    for (int r = 0; r < 8; ++r) { m[r] = -INFINITY; l[r] = 0.0f; }

    bf16_t* Pw = &Pb[wave * 16 * LDP];

    for (int kt = 0; kt < NT; ++kt) {
        const int buf = kt & 1;
        __syncthreads();                       // prev iteration done reading buf^1
        if (kt + 1 < NT) issue_tile(kt + 1, buf ^ 1);
#if HAVE_ASYNC
        // Async loads retire in order: <=8 outstanding means tile kt has landed.
        if (kt + 1 < NT) __builtin_amdgcn_s_wait_asynccnt(8);
        else             __builtin_amdgcn_s_wait_asynccnt(0);
#endif
        __syncthreads();                       // tile kt visible to all waves

        const bf16_t* Kt = KtS[buf];
        const bf16_t* Vt = VtS[buf];

        // S = Q * K^T : 16x64 per wave (4 n-tiles x 2 k-chunks of WMMA).
        v8f s[4];
#pragma unroll
        for (int j = 0; j < 4; ++j) {
            v8f acc;
#pragma unroll
            for (int r = 0; r < 8; ++r) acc[r] = 0.0f;
#pragma unroll
            for (int c = 0; c < 2; ++c) {
                v16bf bk = lds_chunk(&Kt[(j * 16 + ln) * LDT], c * 32, hi);
                acc = __builtin_amdgcn_wmma_f32_16x16x32_bf16(
                        false, aq[c], false, bk, (short)0, acc, false, false);
            }
            s[j] = acc;
        }

        // Scale + reference's bias: subtract NEG_BIAS where (vmask && k<=q).
#pragma unroll
        for (int j = 0; j < 4; ++j) {
            int col = kt * BK + j * 16 + ln;
            bool kv = vmask[(long)b * SK + col] != 0;
#pragma unroll
            for (int r = 0; r < 8; ++r) {
                int rowg = q0 + r + 8 * hi;
                float bias = (kv && (col <= rowg)) ? NEGB : 0.0f;
                s[j][r] = s[j][r] * SCALE - bias;
            }
        }

        // Online softmax (rows live across 16-lane groups).
        float mn[8], alpha[8], rs[8];
#pragma unroll
        for (int r = 0; r < 8; ++r) {
            float mx = fmaxf(fmaxf(s[0][r], s[1][r]), fmaxf(s[2][r], s[3][r]));
#pragma unroll
            for (int off = 1; off < 16; off <<= 1)
                mx = fmaxf(mx, __shfl_xor(mx, off, 16));
            mn[r]    = fmaxf(m[r], mx);
            alpha[r] = __expf(m[r] - mn[r]);
            m[r]     = mn[r];
            rs[r]    = 0.0f;
        }
#pragma unroll
        for (int j = 0; j < 4; ++j)
#pragma unroll
            for (int r = 0; r < 8; ++r) {
                float p = __expf(s[j][r] - mn[r]);
                s[j][r] = p;
                rs[r] += p;
            }
#pragma unroll
        for (int r = 0; r < 8; ++r) {
#pragma unroll
            for (int off = 1; off < 16; off <<= 1)
                rs[r] += __shfl_xor(rs[r], off, 16);
            l[r] = l[r] * alpha[r] + rs[r];
        }
#pragma unroll
        for (int dk = 0; dk < 4; ++dk)
#pragma unroll
            for (int r = 0; r < 8; ++r) o[dk][r] *= alpha[r];

        // D-layout -> A-layout via per-wave LDS staging (bf16).
#pragma unroll
        for (int j = 0; j < 4; ++j)
#pragma unroll
            for (int r = 0; r < 8; ++r)
                Pw[(r + 8 * hi) * LDP + j * 16 + ln] = (bf16_t)s[j][r];
        asm volatile("s_wait_dscnt 0" ::: "memory");  // wave-local store->load fence

        // O += P * V
#pragma unroll
        for (int dk = 0; dk < 4; ++dk) {
#pragma unroll
            for (int c = 0; c < 2; ++c) {
                v16bf ap = lds_chunk(&Pw[ln * LDP], c * 32, hi);
                v16bf bv = lds_chunk(&Vt[(dk * 16 + ln) * LDT], c * 32, hi);
                o[dk] = __builtin_amdgcn_wmma_f32_16x16x32_bf16(
                          false, ap, false, bv, (short)0, o[dk], false, false);
            }
        }
    }

    // Epilogue: normalize by l, apply queries_mask, store f32.
#pragma unroll
    for (int r = 0; r < 8; ++r) {
        int rowg = q0 + r + 8 * hi;
        float inv = (qmask[(long)b * SQ + rowg] != 0) ? (1.0f / l[r]) : 0.0f;
#pragma unroll
        for (int dk = 0; dk < 4; ++dk)
            Out[((long)b * SQ + rowg) * CD + h * DH + dk * 16 + ln] = o[dk][r] * inv;
    }
}

// ---------------------------------------------------------------------------
// Fallback: round-1 style kernel reading f32 directly (used if ws too small).
// ---------------------------------------------------------------------------
__global__ __launch_bounds__(128)
void fattn_wmma_f32(const float* __restrict__ Q, const float* __restrict__ K,
                    const float* __restrict__ V,
                    const unsigned char* __restrict__ qmask,
                    const unsigned char* __restrict__ vmask,
                    float* __restrict__ Out) {
    __shared__ __align__(16) bf16_t Kt[BK * LDK];
    __shared__ __align__(16) bf16_t Vt[DH * LDV];
    __shared__ __align__(16) bf16_t Pb[NW * 16 * LDP];

    const int qt = blockIdx.x, h = blockIdx.y, b = blockIdx.z, tid = threadIdx.x;
    const int wave = tid >> 5, lane = tid & 31, ln = lane & 15, hi = lane >> 4;
    const int q0 = qt * BQ + wave * 16;

    const long qoff = ((long)b * SQ + (q0 + ln)) * CD + h * DH;
    v16bf aq[2];
    aq[0] = gmem_f32_chunk(Q + qoff, 0, hi);
    aq[1] = gmem_f32_chunk(Q + qoff, 32, hi);

    v8f o[4];
    float m[8], l[8];
#pragma unroll
    for (int dk = 0; dk < 4; ++dk)
#pragma unroll
        for (int r = 0; r < 8; ++r) o[dk][r] = 0.0f;
#pragma unroll
    for (int r = 0; r < 8; ++r) { m[r] = -INFINITY; l[r] = 0.0f; }

    const long kvbase = (long)b * SK * CD + h * DH;
    bf16_t* Pw = &Pb[wave * 16 * LDP];

    for (int kt = 0; kt < NT; ++kt) {
        __syncthreads();
        const float* Kg = K + kvbase + (long)kt * BK * CD;
        const float* Vg = V + kvbase + (long)kt * BK * CD;
#pragma unroll
        for (int i = 0; i < 8; ++i) {
            int chunk = i * 128 + tid;
            int r = chunk >> 4, c4 = (chunk & 15) * 4;
            v4f kd = *reinterpret_cast<const v4f*>(Kg + (long)r * CD + c4);
            v4f vd = *reinterpret_cast<const v4f*>(Vg + (long)r * CD + c4);
#pragma unroll
            for (int t = 0; t < 4; ++t) {
                Kt[r * LDK + c4 + t]   = (bf16_t)kd[t];
                Vt[(c4 + t) * LDV + r] = (bf16_t)vd[t];
            }
        }
        __syncthreads();

        v8f s[4];
#pragma unroll
        for (int j = 0; j < 4; ++j) {
            v8f acc;
#pragma unroll
            for (int r = 0; r < 8; ++r) acc[r] = 0.0f;
#pragma unroll
            for (int c = 0; c < 2; ++c) {
                v16bf bk = lds_chunk(&Kt[(j * 16 + ln) * LDK], c * 32, hi);
                acc = __builtin_amdgcn_wmma_f32_16x16x32_bf16(
                        false, aq[c], false, bk, (short)0, acc, false, false);
            }
            s[j] = acc;
        }
#pragma unroll
        for (int j = 0; j < 4; ++j) {
            int col = kt * BK + j * 16 + ln;
            bool kv = vmask[(long)b * SK + col] != 0;
#pragma unroll
            for (int r = 0; r < 8; ++r) {
                int rowg = q0 + r + 8 * hi;
                float bias = (kv && (col <= rowg)) ? NEGB : 0.0f;
                s[j][r] = s[j][r] * SCALE - bias;
            }
        }
        float mn[8], alpha[8], rs[8];
#pragma unroll
        for (int r = 0; r < 8; ++r) {
            float mx = fmaxf(fmaxf(s[0][r], s[1][r]), fmaxf(s[2][r], s[3][r]));
#pragma unroll
            for (int off = 1; off < 16; off <<= 1)
                mx = fmaxf(mx, __shfl_xor(mx, off, 16));
            mn[r] = fmaxf(m[r], mx);
            alpha[r] = __expf(m[r] - mn[r]);
            m[r] = mn[r];
            rs[r] = 0.0f;
        }
#pragma unroll
        for (int j = 0; j < 4; ++j)
#pragma unroll
            for (int r = 0; r < 8; ++r) {
                float p = __expf(s[j][r] - mn[r]);
                s[j][r] = p;
                rs[r] += p;
            }
#pragma unroll
        for (int r = 0; r < 8; ++r) {
#pragma unroll
            for (int off = 1; off < 16; off <<= 1)
                rs[r] += __shfl_xor(rs[r], off, 16);
            l[r] = l[r] * alpha[r] + rs[r];
        }
#pragma unroll
        for (int dk = 0; dk < 4; ++dk)
#pragma unroll
            for (int r = 0; r < 8; ++r) o[dk][r] *= alpha[r];
#pragma unroll
        for (int j = 0; j < 4; ++j)
#pragma unroll
            for (int r = 0; r < 8; ++r)
                Pw[(r + 8 * hi) * LDP + j * 16 + ln] = (bf16_t)s[j][r];
        asm volatile("s_wait_dscnt 0" ::: "memory");
#pragma unroll
        for (int dk = 0; dk < 4; ++dk) {
#pragma unroll
            for (int c = 0; c < 2; ++c) {
                v16bf ap = lds_chunk(&Pw[ln * LDP], c * 32, hi);
                v16bf bv = lds_chunk(&Vt[(dk * 16 + ln) * LDV], c * 32, hi);
                o[dk] = __builtin_amdgcn_wmma_f32_16x16x32_bf16(
                          false, ap, false, bv, (short)0, o[dk], false, false);
            }
        }
    }
#pragma unroll
    for (int r = 0; r < 8; ++r) {
        int rowg = q0 + r + 8 * hi;
        float inv = (qmask[(long)b * SQ + rowg] != 0) ? (1.0f / l[r]) : 0.0f;
#pragma unroll
        for (int dk = 0; dk < 4; ++dk)
            Out[((long)b * SQ + rowg) * CD + h * DH + dk * 16 + ln] = o[dk][r] * inv;
    }
}

extern "C" void kernel_launch(void* const* d_in, const int* in_sizes, int n_in,
                              void* d_out, int out_size, void* d_ws, size_t ws_size,
                              hipStream_t stream) {
    (void)in_sizes; (void)n_in; (void)out_size;
    const float* Q = (const float*)d_in[0];
    const float* K = (const float*)d_in[1];
    const float* V = (const float*)d_in[2];
    const unsigned char* qm = (const unsigned char*)d_in[3];
    const unsigned char* vm = (const unsigned char*)d_in[4];
    float* Out = (float*)d_out;

    const size_t nK  = (size_t)2 * SK * CD;            // elements
    const size_t nV  = (size_t)2 * NHEADS * DH * SK;   // elements
    const size_t need = (nK + nV) * sizeof(bf16_t);

    dim3 grid(SQ / BQ, NHEADS, 2);
    dim3 block(NW * 32);

    if (ws_size >= need) {
        bf16_t* Kb  = (bf16_t*)d_ws;
        bf16_t* Vtw = Kb + nK;
        hipLaunchKernelGGL(convert_k_bf16, dim3(nK / (256 * 8)), dim3(256), 0, stream, K, Kb);
        hipLaunchKernelGGL(transpose_v_bf16, dim3(SK / BK, NHEADS, 2), dim3(128), 0, stream, V, Vtw);
        hipLaunchKernelGGL(fattn_wmma_bf16, grid, block, 0, stream, Q, Kb, Vtw, qm, vm, Out);
    } else {
        hipLaunchKernelGGL(fattn_wmma_f32, grid, block, 0, stream, Q, K, V, qm, vm, Out);
    }
}
